// IHNNet_48919677501864
// MI455X (gfx1250) — matrix operand order repaired and checked by reading, not compiled, not run on hardware
//
#include <hip/hip_runtime.h>

typedef _Float16 h16;
typedef __attribute__((ext_vector_type(16))) _Float16 v16h;
typedef __attribute__((ext_vector_type(8)))  _Float16 v8h;
typedef __attribute__((ext_vector_type(8)))  float    v8f;

#define NA_PRERELU 1
#define NA_RES     2
#define NA_RELU    4

// ---------------------------------------------------------------- small utils
__global__ void k_zero(float* p, int n) {
  int i = blockIdx.x * blockDim.x + threadIdx.x;
  if (i < n) p[i] = 0.f;
}

__global__ void k_cast(const float* __restrict__ s, h16* __restrict__ d, int n) {
  int i = blockIdx.x * blockDim.x + threadIdx.x;
  if (i < n) d[i] = (h16)s[i];
}

// NCHW f32 image (B,3,H,W) -> NHWC f16, C padded to 32, val = 2x-1
__global__ void k_pack(const float* __restrict__ src, h16* __restrict__ dst,
                       int B, int H, int W) {
  int idx = blockIdx.x * blockDim.x + threadIdx.x;
  int total = B * H * W * 32;
  if (idx >= total) return;
  int c = idx & 31;
  int p = idx >> 5;
  float v = 0.f;
  if (c < 3) {
    int x = p % W; int t = p / W; int y = t % H; int b = t / H;
    v = 2.f * src[(((size_t)b * 3 + c) * H + y) * W + x] - 1.f;
  }
  dst[idx] = (h16)v;
}

// OIHW f32 -> [tap][co][ci_pad] f16 (zero pad ci)
__global__ void k_wtrans(const float* __restrict__ w, h16* __restrict__ out,
                         int Co, int Ci, int KH, int KW, int Cip) {
  int idx = blockIdx.x * blockDim.x + threadIdx.x;
  int total = KH * KW * Co * Cip;
  if (idx >= total) return;
  int cj = idx % Cip;
  int t = idx / Cip;
  int co = t % Co;
  int tap = t / Co;
  int ky = tap / KW, kx = tap - ky * KW;
  float v = (cj < Ci) ? w[(((size_t)co * Ci + cj) * KH + ky) * KW + kx] : 0.f;
  out[idx] = (h16)v;
}

// ---------------------------------------------------------------- WMMA tile core
// LDS tiles use row stride 40 halves (80B) -> conflict-free b128 fragment reads.
// A fragment (16x32 f16, ISA 7.12.2): lane m=l%16, k0=(l<16?0:8), halves [k0..k0+7],[k0+16..k0+23]
// B fragment (32x16 f16): lane n=l%16, k0=(l<16?0:8), halves [k0..k0+7],[k0+16..k0+23]
__device__ __forceinline__
v8f wmma_tile(const h16* As, const h16* Bs, int lane, int wm, int wn, v8f acc) {
  const int mrow = wm * 16 + (lane & 15);
  const int ka = (lane < 16) ? 0 : 8;
  const int nrow = wn * 16 + (lane & 15);
  const int kb = (lane < 16) ? 0 : 8;
  v8h alo = *(const v8h*)(As + mrow * 40 + ka);
  v8h ahi = *(const v8h*)(As + mrow * 40 + ka + 16);
  v8h blo = *(const v8h*)(Bs + nrow * 40 + kb);
  v8h bhi = *(const v8h*)(Bs + nrow * 40 + kb + 16);
  v16h a, b;
#pragma unroll
  for (int i = 0; i < 8; ++i) {
    a[i] = alo[i]; a[i + 8] = ahi[i];
    b[i] = blo[i]; b[i + 8] = bhi[i];
  }
  return __builtin_amdgcn_wmma_f32_16x16x32_f16(false, a, false, b, (short)0, acc,
                                                false, false);
}

// ---------------------------------------------------------------- implicit-GEMM conv
// X: f16 NHWC [B][H][W][Cip]; Wt: f16 [KH*KW][Co][Cip]; Y: f32 [B*H*W][Co]
// M = B*H*W multiple of 64; Co multiple of 32; Cip multiple of 32.
// Halo taps need zero-fill, so this kernel stages through VGPRs -> LDS.
__global__ __launch_bounds__(256)
void k_conv_wmma(const h16* __restrict__ X, const h16* __restrict__ Wt,
                 const float* __restrict__ bias, float* __restrict__ Y,
                 int H, int W, int Cip, int Co, int KH, int KW, int Pd) {
  __shared__ h16 As[64 * 40];
  __shared__ h16 Bs[32 * 40];
  const int tid = threadIdx.x;
  const int lane = tid & 31;
  const int wv = tid >> 5;
  const int wm = wv & 3;   // 0..3  -> m tile
  const int wn = wv >> 2;  // 0..1  -> n tile
  const int m0 = blockIdx.x * 64;
  const int n0 = blockIdx.y * 32;

  // A staging: 64 rows x 32 halves, 4 threads/row x 8 halves
  const int pa = tid >> 2;
  const int ca = (tid & 3) << 3;
  const int gpos = m0 + pa;
  const int xx = gpos % W;
  int t2 = gpos / W;
  const int yy = t2 % H;
  const int bb = t2 / H;

  // B staging: 32 rows x 32 halves, 8 threads/row x 4 halves
  const int pb = tid >> 3;
  const int cb = (tid & 7) << 2;

  v8f acc = {};
  const int taps = KH * KW;
  for (int tap = 0; tap < taps; ++tap) {
    const int ky = tap / KW, kx = tap - ky * KW;
    const int sy = yy + ky - Pd, sx = xx + kx - Pd;
    const bool inb = (sy >= 0) && (sy < H) && (sx >= 0) && (sx < W);
    const h16* xrow = X + (((size_t)bb * H + sy) * W + sx) * Cip;
    const h16* wrow = Wt + ((size_t)tap * Co + (n0 + pb)) * Cip;
    for (int kc = 0; kc < Cip; kc += 32) {
      __syncthreads();
      v8h av = {};
      if (inb) av = *(const v8h*)(xrow + kc + ca);
      *(v8h*)&As[pa * 40 + ca] = av;
      *(float2*)&Bs[pb * 40 + cb] = *(const float2*)(wrow + kc + cb);
      if (kc + 32 < Cip) __builtin_prefetch(wrow + kc + 32 + cb, 0, 1);
      __syncthreads();
      acc = wmma_tile(As, Bs, lane, wm, wn, acc);
    }
  }

  const int nco = n0 + wn * 16 + (lane & 15);
  const int mbase = m0 + wm * 16 + ((lane < 16) ? 0 : 8);
  const float bv = bias ? bias[nco] : 0.f;
#pragma unroll
  for (int r = 0; r < 8; ++r)
    Y[(size_t)(mbase + r) * Co + nco] = acc[r] + bv;
}

// ---------------------------------------------------------------- correlation GEMM
// c0[b][pq][pm] = scale * sum_c FQ[b][pq][c] * FK[b][pm][c]
// No boundary conditions -> stage tiles with GLOBAL_LOAD_ASYNC_TO_LDS_B128
// (ASYNCcnt path, ISA 10.x/15.18), one b128 per lane, then s_wait_asynccnt + barrier.
__global__ __launch_bounds__(256)
void k_corr_wmma(const h16* __restrict__ FQ, const h16* __restrict__ FK,
                 float* __restrict__ Cv, int HWp, int C, float scale) {
  __shared__ h16 As[64 * 40];
  __shared__ h16 Bs[32 * 40];
  const int tid = threadIdx.x;
  const int lane = tid & 31;
  const int wv = tid >> 5;
  const int wm = wv & 3;
  const int wn = wv >> 2;
  const int m0 = blockIdx.x * 64;
  const int n0 = blockIdx.y * 32;
  const int b = blockIdx.z;
  const h16* Ab = FQ + (size_t)b * HWp * C;
  const h16* Bb = FK + (size_t)b * HWp * C;

  // A: 64 rows x 32 halves, 4 lanes/row x 8 halves (16B) -> 256 async b128
  const int pa = tid >> 2;
  const int ca = (tid & 3) << 3;
  const unsigned ldsA = (unsigned)(uintptr_t)&As[pa * 40 + ca];
  // B: 32 rows x 32 halves, 4 lanes/row x 8 halves -> threads 0..127 async b128
  const int pbb = tid >> 2;          // valid for tid < 128 -> 0..31
  const int cbb = (tid & 3) << 3;
  const unsigned ldsB = (unsigned)(uintptr_t)&Bs[(pbb & 31) * 40 + cbb];

  v8f acc = {};
  for (int kc = 0; kc < C; kc += 32) {
    __syncthreads();
    {
      const h16* ga = Ab + (size_t)(m0 + pa) * C + kc + ca;
      asm volatile("global_load_async_to_lds_b128 %0, %1, off"
                   :: "v"(ldsA), "v"((unsigned long long)(uintptr_t)ga)
                   : "memory");
      if (tid < 128) {
        const h16* gb = Bb + (size_t)(n0 + pbb) * C + kc + cbb;
        asm volatile("global_load_async_to_lds_b128 %0, %1, off"
                     :: "v"(ldsB), "v"((unsigned long long)(uintptr_t)gb)
                     : "memory");
      }
      asm volatile("s_wait_asynccnt 0" ::: "memory");
    }
    __syncthreads();
    acc = wmma_tile(As, Bs, lane, wm, wn, acc);
  }
  const int nco = n0 + wn * 16 + (lane & 15);
  const int mbase = m0 + wm * 16 + ((lane < 16) ? 0 : 8);
#pragma unroll
  for (int r = 0; r < 8; ++r)
    Cv[((size_t)b * HWp + (mbase + r)) * HWp + nco] = acc[r] * scale;
}

// c1[b][p][U][V] = avg of 2x2 block of c0[b][p][:][:]   (n = 16*1024*256)
__global__ void k_corrpool(const float* __restrict__ c0, float* __restrict__ c1, int n) {
  int idx = blockIdx.x * blockDim.x + threadIdx.x;
  if (idx >= n) return;
  int V = idx & 15; int t = idx >> 4;
  int U = t & 15; t >>= 4;
  int p = t & 1023; int b = t >> 10;
  const float* s = c0 + (((size_t)b * 1024 + p) << 10);
  int u = U * 2, v = V * 2;
  c1[idx] = 0.25f * (s[u * 32 + v] + s[u * 32 + v + 1] +
                     s[(u + 1) * 32 + v] + s[(u + 1) * 32 + v + 1]);
}

// ---------------------------------------------------------------- norm statistics
// one block per (b, group); src f32 [B][HW][C]; stats[2*gid] = mean, [2*gid+1] = rstd
__global__ __launch_bounds__(256)
void k_stats(const float* __restrict__ src, float* __restrict__ stats,
             int HW, int C, int cpg) {
  const int groups = C / cpg;
  const int b = blockIdx.x / groups;
  const int g = blockIdx.x - b * groups;
  const float* base = src + (size_t)b * HW * C + g * cpg;
  const int n = HW * cpg;
  float s = 0.f, ss = 0.f;
  for (int i = threadIdx.x; i < n; i += 256) {
    int p = i / cpg, j = i - p * cpg;
    float v = base[(size_t)p * C + j];
    s += v; ss += v * v;
  }
  __shared__ float rs[256], rq[256];
  rs[threadIdx.x] = s; rq[threadIdx.x] = ss;
  __syncthreads();
  for (int k = 128; k > 0; k >>= 1) {
    if (threadIdx.x < k) { rs[threadIdx.x] += rs[threadIdx.x + k];
                           rq[threadIdx.x] += rq[threadIdx.x + k]; }
    __syncthreads();
  }
  if (threadIdx.x == 0) {
    float m = rs[0] / (float)n;
    float var = rq[0] / (float)n - m * m;
    stats[2 * blockIdx.x] = m;
    stats[2 * blockIdx.x + 1] = rsqrtf(var + 1e-5f);
  }
}

// normalize (+affine) (+prerelu) (+residual) (+relu) -> f16 NHWC
__global__ void k_normact(const float* __restrict__ src, const float* __restrict__ stats,
                          const float* __restrict__ gamma, const float* __restrict__ beta,
                          const h16* __restrict__ res, h16* __restrict__ dst,
                          int HW, int C, int cpg, int flags, int total) {
  int idx = blockIdx.x * blockDim.x + threadIdx.x;
  if (idx >= total) return;
  int c = idx % C;
  int bp = idx / C;
  int b = bp / HW;
  int groups = C / cpg;
  int g = c / cpg;
  float m = stats[2 * (b * groups + g)];
  float r = stats[2 * (b * groups + g) + 1];
  float v = (src[idx] - m) * r;
  if (gamma) v = v * gamma[c] + beta[c];
  if (flags & NA_PRERELU) v = fmaxf(v, 0.f);
  if (res) v += (float)res[idx];
  if (flags & NA_RELU) v = fmaxf(v, 0.f);
  dst[idx] = (h16)v;
}

// maxpool 2x2 stride 2, f16 NHWC
__global__ void k_pool(const h16* __restrict__ src, h16* __restrict__ dst,
                       int H, int W, int C, int total) {
  int idx = blockIdx.x * blockDim.x + threadIdx.x;
  if (idx >= total) return;
  int c = idx % C; int t = idx / C;
  int Wo = W >> 1, Ho = H >> 1;
  int xo = t % Wo; t /= Wo;
  int yo = t % Ho; int b = t / Ho;
  const h16* s = src + (((size_t)b * H + 2 * yo) * W + 2 * xo) * C + c;
  float a0 = (float)s[0], a1 = (float)s[C];
  float a2 = (float)s[(size_t)W * C], a3 = (float)s[(size_t)W * C + C];
  dst[idx] = (h16)fmaxf(fmaxf(a0, a1), fmaxf(a2, a3));
}

// ---------------------------------------------------------------- 4-pt DLT (8x8 solve)
__global__ void k_homography(const float* __restrict__ disps, float scale,
                             float cw, float ch, float* __restrict__ Hout, int Bn) {
  int b = blockIdx.x * blockDim.x + threadIdx.x;
  if (b >= Bn) return;
  const float sx[4] = {0.f, cw, 0.f, cw};
  const float sy[4] = {0.f, 0.f, ch, ch};
  float A[8][9];
  for (int i = 0; i < 4; ++i) {
    float x = sx[i], y = sy[i];
    float u = x + disps[(b * 4 + i) * 2 + 0] * scale;
    float v = y + disps[(b * 4 + i) * 2 + 1] * scale;
    float* r0 = A[i];
    r0[0] = x; r0[1] = y; r0[2] = 1.f; r0[3] = 0.f; r0[4] = 0.f; r0[5] = 0.f;
    r0[6] = -u * x; r0[7] = -u * y; r0[8] = u;
    float* r1 = A[4 + i];
    r1[0] = 0.f; r1[1] = 0.f; r1[2] = 0.f; r1[3] = x; r1[4] = y; r1[5] = 1.f;
    r1[6] = -v * x; r1[7] = -v * y; r1[8] = v;
  }
  for (int k = 0; k < 8; ++k) {
    int piv = k; float mx = fabsf(A[k][k]);
    for (int r = k + 1; r < 8; ++r) {
      float t = fabsf(A[r][k]);
      if (t > mx) { mx = t; piv = r; }
    }
    if (piv != k)
      for (int c = k; c < 9; ++c) { float t = A[k][c]; A[k][c] = A[piv][c]; A[piv][c] = t; }
    float inv = 1.f / A[k][k];
    for (int c = k; c < 9; ++c) A[k][c] *= inv;
    for (int r = 0; r < 8; ++r) {
      if (r == k) continue;
      float f = A[r][k];
      for (int c = k; c < 9; ++c) A[r][c] -= f * A[k][c];
    }
  }
  float* H = Hout + b * 9;
  for (int k = 0; k < 8; ++k) H[k] = A[k][8];
  H[8] = 1.f;
}

// ---------------------------------------------------------------- GMA input builder
// out f16 [B][1024][192]: ch 0..80 level0 corr, 81..161 level1 corr, 162/163 flow, rest 0
__global__ void k_gmain(const float* __restrict__ Hm, const float* __restrict__ c0,
                        const float* __restrict__ c1, h16* __restrict__ out) {
  int idx = blockIdx.x * blockDim.x + threadIdx.x;
  if (idx >= 16 * 1024 * 192) return;
  int ch = idx % 192;
  int p = (idx / 192) & 1023;
  int b = idx / (192 * 1024);
  float gx = (float)(p & 31);
  float gy = (float)(p >> 5);
  const float* H = Hm + b * 9;
  float wz = H[6] * gx + H[7] * gy + H[8];
  float wx = (H[0] * gx + H[1] * gy + H[2]) / wz;
  float wy = (H[3] * gx + H[4] * gy + H[5]) / wz;
  float val = 0.f;
  if (ch < 162) {
    int l = ch / 81, k = ch - l * 81;
    float dx = (float)(k % 9) - 4.f;
    float dy = (float)(k / 9) - 4.f;
    const float* slice;
    float px, py;
    int Hl;
    if (l == 0) { px = wx + dx; py = wy + dy; Hl = 32;
                  slice = c0 + (((size_t)b * 1024 + p) << 10); }
    else        { px = wx * 0.5f + dx; py = wy * 0.5f + dy; Hl = 16;
                  slice = c1 + (((size_t)b * 1024 + p) << 8); }
    float x0 = floorf(px), y0 = floorf(py);
    float fx = px - x0, fy = py - y0;
    int xi = (int)x0, yi = (int)y0;
    float v00 = 0.f, v01 = 0.f, v10 = 0.f, v11 = 0.f;
    if (yi >= 0 && yi < Hl) {
      if (xi >= 0 && xi < Hl)         v00 = slice[yi * Hl + xi];
      if (xi + 1 >= 0 && xi + 1 < Hl) v01 = slice[yi * Hl + xi + 1];
    }
    if (yi + 1 >= 0 && yi + 1 < Hl) {
      if (xi >= 0 && xi < Hl)         v10 = slice[(yi + 1) * Hl + xi];
      if (xi + 1 >= 0 && xi + 1 < Hl) v11 = slice[(yi + 1) * Hl + xi + 1];
    }
    val = v00 * (1.f - fx) * (1.f - fy) + v01 * fx * (1.f - fy)
        + v10 * (1.f - fx) * fy + v11 * fx * fy;
  } else if (ch == 162) {
    val = wx - gx;
  } else if (ch == 163) {
    val = wy - gy;
  }
  out[idx] = (h16)val;
}

// ---------------------------------------------------------------- final 1x1 head + update
// a: f16 [16][4][128] (2x2 spatial NHWC); w f32 [2][128]; disps += head(a); lst = disps
__global__ void k_g10b(const h16* __restrict__ a, const float* __restrict__ w,
                       const float* __restrict__ bias, float* __restrict__ disps,
                       float* __restrict__ out_lst) {
  int t = blockIdx.x * blockDim.x + threadIdx.x;
  if (t >= 128) return;
  int b = t >> 3;
  int j = (t >> 1) & 3;   // spatial pos
  int i = t & 1;          // channel
  const h16* av = a + ((size_t)b * 4 + j) * 128;
  const float* wv = w + i * 128;
  float s = bias[i];
  for (int c = 0; c < 128; ++c) s += (float)av[c] * wv[c];
  int o = (b * 4 + j) * 2 + i;
  float nd = disps[o] + s;
  disps[o] = nd;
  out_lst[o] = nd;
}

// ---------------------------------------------------------------- host orchestration
enum {
  I_MOV = 0, I_FIX,
  P_STEM_W, P_STEM_B,
  P_RB1_W1, P_RB1_B1, P_RB1_W2, P_RB1_B2,
  P_RB2_W1, P_RB2_B1, P_RB2_W2, P_RB2_B2,
  P_RB3_W1, P_RB3_B1, P_RB3_W2, P_RB3_B2, P_RB3_WD, P_RB3_BD,
  P_RB4_W1, P_RB4_B1, P_RB4_W2, P_RB4_B2,
  P_HEAD1_W, P_HEAD1_B, P_HEAD2_W, P_HEAD2_B,
  P_G1_W, P_G1_B, P_G1_GW, P_G1_GB,
  P_G2_W, P_G2_B, P_G2_GW, P_G2_GB,
  P_G3_W, P_G3_B, P_G3_GW, P_G3_GB,
  P_G4_W, P_G4_B, P_G4_GW, P_G4_GB,
  P_G10A_W, P_G10A_B, P_G10A_GW, P_G10A_GB,
  P_G10B_W, P_G10B_B
};

extern "C" void kernel_launch(void* const* d_in, const int* in_sizes, int n_in,
                              void* d_out, int out_size, void* d_ws, size_t ws_size,
                              hipStream_t stream) {
  (void)in_sizes; (void)n_in; (void)out_size; (void)ws_size;
  const int B = 16;
  auto fin = [&](int i) { return (const float*)d_in[i]; };
  float* dout = (float*)d_out;

  char* base = (char*)d_ws;
  size_t off = 0;
  auto alloc = [&](size_t bytes) -> void* {
    void* p = base + off;
    off = (off + bytes + 255) & ~(size_t)255;
    return p;
  };
  auto allocH = [&](size_t n) { return (h16*)alloc(n * sizeof(h16)); };
  auto allocF = [&](size_t n) { return (float*)alloc(n * sizeof(float)); };

  // ---- weight transforms (device side, every call)
  auto wt = [&](int idx, int Co, int Ci, int K) -> h16* {
    int Cip = (Ci + 31) & ~31;
    h16* buf = allocH((size_t)K * K * Co * Cip);
    int n = K * K * Co * Cip;
    k_wtrans<<<(n + 255) / 256, 256, 0, stream>>>(fin(idx), buf, Co, Ci, K, K, Cip);
    return buf;
  };
  h16* W_stem = wt(P_STEM_W, 64, 3, 7);
  h16* W_rb1a = wt(P_RB1_W1, 64, 64, 3);
  h16* W_rb1b = wt(P_RB1_W2, 64, 64, 3);
  h16* W_rb2a = wt(P_RB2_W1, 64, 64, 3);
  h16* W_rb2b = wt(P_RB2_W2, 64, 64, 3);
  h16* W_rb3a = wt(P_RB3_W1, 96, 64, 3);
  h16* W_rb3b = wt(P_RB3_W2, 96, 96, 3);
  h16* W_rb3d = wt(P_RB3_WD, 96, 64, 1);
  h16* W_rb4a = wt(P_RB4_W1, 96, 96, 3);
  h16* W_rb4b = wt(P_RB4_W2, 96, 96, 3);
  h16* W_h2   = wt(P_HEAD2_W, 256, 96, 1);
  h16* W_g1   = wt(P_G1_W, 128, 164, 3);
  h16* W_g2   = wt(P_G2_W, 128, 128, 3);
  h16* W_g3   = wt(P_G3_W, 128, 128, 3);
  h16* W_g4   = wt(P_G4_W, 128, 128, 3);
  h16* W_g10a = wt(P_G10A_W, 128, 128, 3);

  // ---- buffers
  h16*   xin   = allocH((size_t)B * 128 * 128 * 32);
  float* convf = allocF((size_t)B * 128 * 128 * 64);   // largest conv f32 out (stem)
  h16*   bufA  = allocH((size_t)B * 128 * 128 * 64);   // largest f16 act (stem)
  h16*   bufB  = allocH((size_t)B * 64 * 64 * 64);
  h16*   bufC  = allocH((size_t)B * 64 * 64 * 64);
  h16*   fm    = allocH((size_t)B * 1024 * 256);
  h16*   ff    = allocH((size_t)B * 1024 * 256);
  float* c0    = allocF((size_t)B * 1024 * 1024);
  float* c1    = allocF((size_t)B * 1024 * 256);
  h16*   gmx   = allocH((size_t)B * 1024 * 192);
  float* stats = allocF(2 * 16 * 256);
  float* Hm    = allocF(B * 9);
  float* disps = allocF(B * 8);

  auto conv = [&](const h16* X, const h16* Wb, const float* bias, float* Y,
                  int H, int W, int Cip, int Co, int K, int Pd) {
    dim3 g((unsigned)((B * H * W) / 64), (unsigned)(Co / 32), 1);
    k_conv_wmma<<<g, 256, 0, stream>>>(X, Wb, bias, Y, H, W, Cip, Co, K, K, Pd);
  };
  auto statsL = [&](const float* src, int HW, int C, int cpg) {
    k_stats<<<B * (C / cpg), 256, 0, stream>>>(src, stats, HW, C, cpg);
  };
  auto normactL = [&](const float* src, const float* gw, const float* gb,
                      const h16* res, h16* dst, int HW, int C, int cpg, int flags) {
    int total = B * HW * C;
    k_normact<<<(total + 255) / 256, 256, 0, stream>>>(src, stats, gw, gb, res, dst,
                                                       HW, C, cpg, flags, total);
  };
  auto poolL = [&](const h16* src, h16* dst, int H, int W, int C) {
    int total = B * (H / 2) * (W / 2) * C;
    k_pool<<<(total + 255) / 256, 256, 0, stream>>>(src, dst, H, W, C, total);
  };

  // ---- feature extractor (head2 branch only; head1 unused by forward)
  auto features = [&](const float* img, h16* feat) {
    int n = B * 128 * 128 * 32;
    k_pack<<<(n + 255) / 256, 256, 0, stream>>>(img, xin, B, 128, 128);
    conv(xin, W_stem, fin(P_STEM_B), convf, 128, 128, 32, 64, 7, 3);
    statsL(convf, 16384, 64, 1);
    normactL(convf, nullptr, nullptr, nullptr, bufA, 16384, 64, 1, NA_RELU);
    poolL(bufA, bufB, 128, 128, 64);                       // 64x64x64, x for rb1
    // rb1
    conv(bufB, W_rb1a, fin(P_RB1_B1), convf, 64, 64, 64, 64, 3, 1);
    statsL(convf, 4096, 64, 1);
    normactL(convf, nullptr, nullptr, nullptr, bufC, 4096, 64, 1, NA_RELU);
    conv(bufC, W_rb1b, fin(P_RB1_B2), convf, 64, 64, 64, 64, 3, 1);
    statsL(convf, 4096, 64, 1);
    normactL(convf, nullptr, nullptr, bufB, (h16*)bufA, 4096, 64, 1,
             NA_PRERELU | NA_RES | NA_RELU);               // bufA = rb1 out
    // rb2
    conv(bufA, W_rb2a, fin(P_RB2_B1), convf, 64, 64, 64, 64, 3, 1);
    statsL(convf, 4096, 64, 1);
    normactL(convf, nullptr, nullptr, nullptr, bufB, 4096, 64, 1, NA_RELU);
    conv(bufB, W_rb2b, fin(P_RB2_B2), convf, 64, 64, 64, 64, 3, 1);
    statsL(convf, 4096, 64, 1);
    normactL(convf, nullptr, nullptr, bufA, bufC, 4096, 64, 1,
             NA_PRERELU | NA_RES | NA_RELU);               // bufC = rb2 out
    poolL(bufC, bufB, 64, 64, 64);                         // 32x32x64, x for rb3
    // rb3 (with 1x1 downsample residual)
    conv(bufB, W_rb3a, fin(P_RB3_B1), convf, 32, 32, 64, 96, 3, 1);
    statsL(convf, 1024, 96, 1);
    normactL(convf, nullptr, nullptr, nullptr, bufA, 1024, 96, 1, NA_RELU);
    conv(bufA, W_rb3b, fin(P_RB3_B2), convf, 32, 32, 96, 96, 3, 1);
    statsL(convf, 1024, 96, 1);
    normactL(convf, nullptr, nullptr, nullptr, bufC, 1024, 96, 1, NA_RELU);  // y2
    conv(bufB, W_rb3d, fin(P_RB3_BD), convf, 32, 32, 64, 96, 1, 0);
    statsL(convf, 1024, 96, 1);
    normactL(convf, nullptr, nullptr, bufC, bufA, 1024, 96, 1,
             NA_RES | NA_RELU);                            // bufA = rb3 out
    // rb4
    conv(bufA, W_rb4a, fin(P_RB4_B1), convf, 32, 32, 96, 96, 3, 1);
    statsL(convf, 1024, 96, 1);
    normactL(convf, nullptr, nullptr, nullptr, bufB, 1024, 96, 1, NA_RELU);
    conv(bufB, W_rb4b, fin(P_RB4_B2), convf, 32, 32, 96, 96, 3, 1);
    statsL(convf, 1024, 96, 1);
    normactL(convf, nullptr, nullptr, bufA, bufC, 1024, 96, 1,
             NA_PRERELU | NA_RES | NA_RELU);               // bufC = rb4 out
    // head2: 1x1 conv 96->256 + bias, no norm
    conv(bufC, W_h2, fin(P_HEAD2_B), convf, 32, 32, 96, 256, 1, 0);
    int nf = B * 1024 * 256;
    k_cast<<<(nf + 255) / 256, 256, 0, stream>>>(convf, feat, nf);
  };

  features((const float*)d_in[I_MOV], fm);
  features((const float*)d_in[I_FIX], ff);

  // ---- correlation volume (batched GEMM, async-LDS staged) + pooled level
  {
    dim3 g(1024 / 64, 1024 / 32, B);
    k_corr_wmma<<<g, 256, 0, stream>>>(ff, fm, c0, 1024, 256, 0.0625f);
    int n = B * 1024 * 256;
    k_corrpool<<<(n + 255) / 256, 256, 0, stream>>>(c0, c1, n);
  }

  k_zero<<<1, 128, 0, stream>>>(disps, B * 8);

  // ---- 6 refinement iterations
  for (int it = 0; it < 6; ++it) {
    k_homography<<<1, 32, 0, stream>>>(disps, 0.25f, 31.f, 31.f, Hm, B);
    int n = B * 1024 * 192;
    k_gmain<<<(n + 255) / 256, 256, 0, stream>>>(Hm, c0, c1, gmx);
    conv(gmx, W_g1, fin(P_G1_B), convf, 32, 32, 192, 128, 3, 1);
    statsL(convf, 1024, 128, 8);
    normactL(convf, fin(P_G1_GW), fin(P_G1_GB), nullptr, bufA, 1024, 128, 8, NA_RELU);
    poolL(bufA, bufB, 32, 32, 128);
    conv(bufB, W_g2, fin(P_G2_B), convf, 16, 16, 128, 128, 3, 1);
    statsL(convf, 256, 128, 8);
    normactL(convf, fin(P_G2_GW), fin(P_G2_GB), nullptr, bufA, 256, 128, 8, NA_RELU);
    poolL(bufA, bufB, 16, 16, 128);
    conv(bufB, W_g3, fin(P_G3_B), convf, 8, 8, 128, 128, 3, 1);
    statsL(convf, 64, 128, 8);
    normactL(convf, fin(P_G3_GW), fin(P_G3_GB), nullptr, bufA, 64, 128, 8, NA_RELU);
    poolL(bufA, bufB, 8, 8, 128);
    conv(bufB, W_g4, fin(P_G4_B), convf, 4, 4, 128, 128, 3, 1);
    statsL(convf, 16, 128, 8);
    normactL(convf, fin(P_G4_GW), fin(P_G4_GB), nullptr, bufA, 16, 128, 8, NA_RELU);
    poolL(bufA, bufB, 4, 4, 128);
    conv(bufB, W_g10a, fin(P_G10A_B), convf, 2, 2, 128, 128, 3, 1);
    statsL(convf, 4, 128, 8);
    normactL(convf, fin(P_G10A_GW), fin(P_G10A_GB), nullptr, bufA, 4, 128, 8, NA_RELU);
    k_g10b<<<1, 128, 0, stream>>>(bufA, fin(P_G10B_W), fin(P_G10B_B), disps,
                                  dout + it * 128);
  }

  // ---- final full-resolution homography -> d_out[768..911]
  k_homography<<<1, 32, 0, stream>>>(disps, 1.f, 127.f, 127.f, dout + 768, B);
}
